// SpaceAwareFrameSampler_27333171871922
// MI455X (gfx1250) — compile-verified, matrix operand order checked by reading, not run on hardware
//
#include <hip/hip_runtime.h>
#include <hip/hip_bf16.h>

// ---------------- problem constants ----------------
#define S_FRAMES   128
#define HW_PX      50176            // 224*224
#define BPF        49               // blocks per frame, 49*1024 = 50176
#define GDIM       48
#define VWORDS     3456             // 48^3 / 32 bits
#define VQUADS     864              // VWORDS / 4
#define KSEL       16
#define RANK_UP    25088            // upper median order statistic (0-indexed)
#define NBINS      4096
#define CAPBUF     2048
#define CONF_THR   0.1f
#define VLAMBDA    20.0f

// ---------------- workspace layout (bytes) ----------------
#define OFF_THR      0                       // 128 f32   per-frame threshold
#define OFF_KEYS     512                     // 6  u32    ordered min/max keys
#define OFF_PARAMS   576                     // 4  f32    pmin.xyz, voxel_size
#define OFF_GAINS    640                     // 128 f32
#define OFF_SELECT   1152                    // 128 i32
#define OFF_COVERED  2048                    // 3456 u32 (16B aligned)
#define OFF_OCC      16384                   // 128*3456 u32 (16B aligned, 13824B/frame)

typedef __attribute__((ext_vector_type(8))) int v8i;

__device__ __forceinline__ unsigned fkey(float f) {
    unsigned u = __float_as_uint(f);
    return u ^ (((unsigned)((int)u >> 31)) | 0x80000000u);   // order-preserving map
}
__device__ __forceinline__ float fdecode(unsigned e) {
    unsigned mask = (e & 0x80000000u) ? 0x80000000u : 0xFFFFFFFFu;
    return __uint_as_float(e ^ mask);
}

// ---------------- K0: zero bitmaps (B128), init min/max keys ----------------
__global__ void k_init(uint4* occ4, int* selected, unsigned* keys) {
    int t = blockIdx.x * blockDim.x + threadIdx.x;
    const int totQ = S_FRAMES * VQUADS + VQUADS;      // occ quads + VQUADS; grid covers occ only
    if (t < totQ) occ4[t] = uint4{0u, 0u, 0u, 0u};    // covered sits right after occ? no:
    if (t < S_FRAMES) selected[t] = 0;
    if (t < 3)  keys[t] = 0xFFFFFFFFu;                // min keys
    if (t >= 3 && t < 6) keys[t] = 0u;                // max keys
}
// covered is at a separate (lower) offset, zero it in its own tiny kernel
__global__ void k_init_cov(uint4* cov4) {
    int t = blockIdx.x * blockDim.x + threadIdx.x;
    if (t < VQUADS) cov4[t] = uint4{0u, 0u, 0u, 0u};
}

// ---------------- K1: per-frame order statistic (rank 25088) ----------------
__global__ __launch_bounds__(256) void k_median(const float* __restrict__ conf,
                                                float* __restrict__ thr) {
    int s = blockIdx.x;
    __shared__ int   hist[NBINS];
    __shared__ float buf[CAPBUF];
    __shared__ int   sBin, sBelow, sCnt;
    const float4* c4 = (const float4*)(conf + (size_t)s * HW_PX);
    const int nq = HW_PX / 4;                          // 12544 float4

    for (int i = threadIdx.x; i < NBINS; i += blockDim.x) hist[i] = 0;
    __syncthreads();
    for (int i = threadIdx.x; i < nq; i += blockDim.x) {
        float4 v = c4[i];
        float vv[4] = {v.x, v.y, v.z, v.w};
        #pragma unroll
        for (int u = 0; u < 4; u++) {
            int b = (int)(vv[u] * (float)NBINS); b = min(max(b, 0), NBINS - 1);
            atomicAdd(&hist[b], 1);
        }
    }
    __syncthreads();
    if (threadIdx.x == 0) {
        int cum = 0, bin = NBINS - 1, below = 0;
        for (int b = 0; b < NBINS; b++) {
            int nb = cum + hist[b];
            if (nb > RANK_UP) { bin = b; below = cum; break; }
            cum = nb;
        }
        sBin = bin; sBelow = below; sCnt = 0;
    }
    __syncthreads();
    int bin = sBin;
    for (int i = threadIdx.x; i < nq; i += blockDim.x) {
        float4 v = c4[i];
        float vv[4] = {v.x, v.y, v.z, v.w};
        #pragma unroll
        for (int u = 0; u < 4; u++) {
            int b = (int)(vv[u] * (float)NBINS); b = min(max(b, 0), NBINS - 1);
            if (b == bin) {
                int p = atomicAdd(&sCnt, 1);
                if (p < CAPBUF) buf[p] = vv[u];
            }
        }
    }
    __syncthreads();
    if (threadIdx.x == 0) {
        int n = min(sCnt, CAPBUF);
        int k = RANK_UP - sBelow;
        if (k >= n) k = n - 1; if (k < 0) k = 0;
        for (int i = 1; i < n; i++) {              // tiny (~12 elems) insertion sort
            float key = buf[i]; int j = i - 1;
            while (j >= 0 && buf[j] > key) { buf[j + 1] = buf[j]; j--; }
            buf[j + 1] = key;
        }
        thr[s] = buf[k];
    }
}

// ---------------- K2: global min/max, 4 points / thread, B128 loads ----------------
__global__ __launch_bounds__(256) void k_minmax(const float* __restrict__ pts,
                                                const float* __restrict__ conf,
                                                const float* __restrict__ thr,
                                                unsigned* keys) {
    __shared__ unsigned smin[3], smax[3];
    if (threadIdx.x < 3) { smin[threadIdx.x] = 0xFFFFFFFFu; smax[threadIdx.x] = 0u; }
    __syncthreads();

    int f = blockIdx.x / BPF;
    int chunk = blockIdx.x % BPF;
    size_t gi = (size_t)f * HW_PX + (size_t)chunk * 1024 + (size_t)threadIdx.x * 4;
    float t = thr[f];

    __builtin_prefetch(pts + 3 * (gi + 4096), 0, 1);   // global_prefetch_b8
    float4 c4 = *(const float4*)(conf + gi);
    const float4* pb = (const float4*)(pts + 3 * gi);  // 48B = 4 points, 16B aligned
    float4 q0 = pb[0];   // x0 y0 z0 x1
    float4 q1 = pb[1];   // y1 z1 x2 y2
    float4 q2 = pb[2];   // z2 x3 y3 z3
    float cc[4] = {c4.x, c4.y, c4.z, c4.w};
    float px[4] = {q0.x, q0.w, q1.z, q2.y};
    float py[4] = {q0.y, q1.x, q1.w, q2.z};
    float pz[4] = {q0.z, q1.y, q2.x, q2.w};

    unsigned lmin[3] = {0xFFFFFFFFu, 0xFFFFFFFFu, 0xFFFFFFFFu};
    unsigned lmax[3] = {0u, 0u, 0u};
    #pragma unroll
    for (int u = 0; u < 4; u++) {
        if (cc[u] > CONF_THR && cc[u] >= t) {
            unsigned kx = fkey(px[u]), ky = fkey(py[u]), kz = fkey(pz[u]);
            lmin[0] = min(lmin[0], kx); lmax[0] = max(lmax[0], kx);
            lmin[1] = min(lmin[1], ky); lmax[1] = max(lmax[1], ky);
            lmin[2] = min(lmin[2], kz); lmax[2] = max(lmax[2], kz);
        }
    }
    #pragma unroll
    for (int a = 0; a < 3; a++) { atomicMin(&smin[a], lmin[a]); atomicMax(&smax[a], lmax[a]); }
    __syncthreads();
    if (threadIdx.x < 3) {
        atomicMin(&keys[threadIdx.x],     smin[threadIdx.x]);
        atomicMax(&keys[3 + threadIdx.x], smax[threadIdx.x]);
    }
}

// ---------------- K3: voxel params ----------------
__global__ void k_params(const unsigned* keys, float* params) {
    if (blockIdx.x == 0 && threadIdx.x == 0) {
        float pmin[3], pmax[3];
        #pragma unroll
        for (int a = 0; a < 3; a++) { pmin[a] = fdecode(keys[a]); pmax[a] = fdecode(keys[3 + a]); }
        float e0 = pmax[0] - pmin[0], e1 = pmax[1] - pmin[1], e2 = pmax[2] - pmin[2];
        float vs = fminf(fminf(e0, e1), e2) / VLAMBDA;
        params[0] = pmin[0]; params[1] = pmin[1]; params[2] = pmin[2]; params[3] = vs;
    }
}

// ---------------- K4: voxel occupancy scatter, 4 points / thread ----------------
__global__ __launch_bounds__(256) void k_occ(const float* __restrict__ pts,
                                             const float* __restrict__ conf,
                                             const float* __restrict__ thr,
                                             const float* __restrict__ params,
                                             unsigned* occ) {
    int f = blockIdx.x / BPF;
    int chunk = blockIdx.x % BPF;
    size_t gi = (size_t)f * HW_PX + (size_t)chunk * 1024 + (size_t)threadIdx.x * 4;
    float t = thr[f];
    float p0 = params[0], p1 = params[1], p2 = params[2], vs = params[3];
    unsigned* o = occ + (size_t)f * VWORDS;

    float4 c4 = *(const float4*)(conf + gi);
    const float4* pb = (const float4*)(pts + 3 * gi);
    float4 q0 = pb[0];
    float4 q1 = pb[1];
    float4 q2 = pb[2];
    float cc[4] = {c4.x, c4.y, c4.z, c4.w};
    float px[4] = {q0.x, q0.w, q1.z, q2.y};
    float py[4] = {q0.y, q1.x, q1.w, q2.z};
    float pz[4] = {q0.z, q1.y, q2.x, q2.w};

    #pragma unroll
    for (int u = 0; u < 4; u++) {
        if (cc[u] > CONF_THR && cc[u] >= t) {
            int ix = (int)floorf((px[u] - p0) / vs);
            int iy = (int)floorf((py[u] - p1) / vs);
            int iz = (int)floorf((pz[u] - p2) / vs);
            ix = min(max(ix, 0), GDIM - 1);
            iy = min(max(iy, 0), GDIM - 1);
            iz = min(max(iz, 0), GDIM - 1);
            int vid = (ix * GDIM + iy) * GDIM + iz;
            atomicOr(&o[vid >> 5], 1u << (vid & 31));
        }
    }
}

// ---------------- K5: greedy gains via WMMA IU8 byte-sum reduction ----------------
// One wave32 per frame. B128 loads: 864 uint4 per frame, 27 per lane, staged 8+8
// quads per chunk so loads clause/pipeline. V_WMMA_I32_16X16X64_IU8 with all-ones
// B row-sums the 16x64-byte A tile; C accumulates; D column N=0 (lanes 0 and 16)
// carries the grand total.
__global__ __launch_bounds__(32) void k_gains(const unsigned* __restrict__ occ,
                                              const unsigned* __restrict__ covered,
                                              const int* __restrict__ selected,
                                              float* __restrict__ gains) {
    int f = blockIdx.x;
    int lane = threadIdx.x;
    if (selected[f]) {                         // wave-uniform branch, no WMMA inside
        if (lane == 0) gains[f] = -1.0f;
        return;
    }
    const uint4* o4 = (const uint4*)(occ + (size_t)f * VWORDS);
    const uint4* c4 = (const uint4*)covered;

    v8i bones = {0x01010101, 0x01010101, 0x01010101, 0x01010101,
                 0x01010101, 0x01010101, 0x01010101, 0x01010101};
    v8i cacc = {0, 0, 0, 0, 0, 0, 0, 0};

    #pragma unroll
    for (int ch = 0; ch < 4; ch++) {
        uint4 oc[8], cv[8];
        #pragma unroll
        for (int jj = 0; jj < 8; jj++) {       // issue all loads before consuming
            int j = ch * 8 + jj;
            if (j < 27) {
                int q = j * 32 + lane;         // coalesced: 512B contiguous per wave
                oc[jj] = o4[q];
                cv[jj] = c4[q];
            } else {
                oc[jj] = uint4{0u, 0u, 0u, 0u};
                cv[jj] = uint4{0u, 0u, 0u, 0u};
            }
        }
        unsigned pk[8];
        #pragma unroll
        for (int jj = 0; jj < 8; jj++) {
            unsigned b0 = (unsigned)__popc(oc[jj].x & ~cv[jj].x);
            unsigned b1 = (unsigned)__popc(oc[jj].y & ~cv[jj].y);
            unsigned b2 = (unsigned)__popc(oc[jj].z & ~cv[jj].z);
            unsigned b3 = (unsigned)__popc(oc[jj].w & ~cv[jj].w);
            pk[jj] = b0 | (b1 << 8) | (b2 << 16) | (b3 << 24);
        }
        v8i a;
        #pragma unroll
        for (int k = 0; k < 8; k++) a[k] = (int)pk[k];
        // (sgn_a, A, sgn_b, B, C, reuse_a, reuse_b)
        cacc = __builtin_amdgcn_wmma_i32_16x16x64_iu8(false, a, false, bones,
                                                      cacc, false, false);
    }
    int partial = cacc[0] + cacc[1] + cacc[2] + cacc[3] +
                  cacc[4] + cacc[5] + cacc[6] + cacc[7];
    int total = __builtin_amdgcn_readlane(partial, 0) +
                __builtin_amdgcn_readlane(partial, 16);        // column N=0 of D
    if (lane == 0) gains[f] = (float)total;
}

// ---------------- K6: argmax (first-max tie-break) + covered update ----------------
__global__ __launch_bounds__(128) void k_select(const unsigned* __restrict__ occ,
                                                unsigned* __restrict__ covered,
                                                int* __restrict__ selected,
                                                const float* __restrict__ gains,
                                                float* __restrict__ out, int iter) {
    __shared__ int sBest;
    if (threadIdx.x == 0) {
        float bg = -2.0f; int bi = 0;
        for (int f = 0; f < S_FRAMES; f++) {
            float g = gains[f];
            if (g > bg) { bg = g; bi = f; }    // strict > => first max (jnp.argmax)
        }
        sBest = bi;
        selected[bi] = 1;
        out[iter]        = (float)bi;          // selected index
        out[KSEL + iter] = bg;                 // coverage score
    }
    __syncthreads();
    const uint4* o4 = (const uint4*)(occ + (size_t)sBest * VWORDS);
    uint4* c4 = (uint4*)covered;
    for (int q = threadIdx.x; q < VQUADS; q += blockDim.x) {
        uint4 o = o4[q], c = c4[q];
        c.x |= o.x; c.y |= o.y; c.z |= o.z; c.w |= o.w;
        c4[q] = c;
    }
}

// ---------------- K7: total coverage popcount ----------------
__global__ __launch_bounds__(256) void k_cov(const unsigned* __restrict__ covered,
                                             float* __restrict__ out) {
    __shared__ int red[256];
    const uint4* c4 = (const uint4*)covered;
    int acc = 0;
    for (int q = threadIdx.x; q < VQUADS; q += 256) {
        uint4 c = c4[q];
        acc += __popc(c.x) + __popc(c.y) + __popc(c.z) + __popc(c.w);
    }
    red[threadIdx.x] = acc;
    __syncthreads();
    for (int s = 128; s > 0; s >>= 1) {
        if (threadIdx.x < s) red[threadIdx.x] += red[threadIdx.x + s];
        __syncthreads();
    }
    if (threadIdx.x == 0) out[2 * KSEL] = (float)red[0];
}

// ---------------- host launcher ----------------
extern "C" void kernel_launch(void* const* d_in, const int* in_sizes, int n_in,
                              void* d_out, int out_size, void* d_ws, size_t ws_size,
                              hipStream_t stream) {
    // inputs (setup_inputs order): depth, depth_conf, world_points, world_points_conf, pose_enc
    const float* wpts  = (const float*)d_in[2];
    const float* wconf = (const float*)d_in[3];
    (void)in_sizes; (void)n_in; (void)out_size; (void)ws_size;

    char* ws = (char*)d_ws;
    float*    thr      = (float*)(ws + OFF_THR);
    unsigned* keys     = (unsigned*)(ws + OFF_KEYS);
    float*    params   = (float*)(ws + OFF_PARAMS);
    float*    gains    = (float*)(ws + OFF_GAINS);
    int*      selected = (int*)(ws + OFF_SELECT);
    unsigned* covered  = (unsigned*)(ws + OFF_COVERED);
    unsigned* occ      = (unsigned*)(ws + OFF_OCC);
    float*    out      = (float*)d_out;

    // zero 110592 occ quads (+ flags/keys) and 864 covered quads
    k_init    <<<(S_FRAMES * VQUADS + 255) / 256, 256, 0, stream>>>((uint4*)occ, selected, keys);
    k_init_cov<<<(VQUADS + 255) / 256, 256, 0, stream>>>((uint4*)covered);
    k_median  <<<S_FRAMES, 256, 0, stream>>>(wconf, thr);
    k_minmax  <<<S_FRAMES * BPF, 256, 0, stream>>>(wpts, wconf, thr, keys);
    k_params  <<<1, 32, 0, stream>>>(keys, params);
    k_occ     <<<S_FRAMES * BPF, 256, 0, stream>>>(wpts, wconf, thr, params, occ);
    for (int it = 0; it < KSEL; it++) {
        k_gains <<<S_FRAMES, 32, 0, stream>>>(occ, covered, selected, gains);
        k_select<<<1, 128, 0, stream>>>(occ, covered, selected, gains, out, it);
    }
    k_cov<<<1, 256, 0, stream>>>(covered, out);
}
